// CMP_83829171683752
// MI455X (gfx1250) — matrix operand order repaired and checked by reading, not compiled, not run on hardware
//
#include <hip/hip_runtime.h>

// ---------------------------------------------------------------------------
// N=1024, C=16, H=W=32, E=4096; conv 48->32->32->16, 3x3 pad 1, leaky 0.1
// Conv strategy: 9 shifted GEMMs (one per tap) via V_WMMA_F32_16X16X32_BF16,
// software-pipelined (double-buffered fragments) so LDS loads for chunk j+1
// overlap the WMMA of chunk j.
// ---------------------------------------------------------------------------
#define NN   1024
#define CC   16
#define HW   1024        // 32*32
#define EE   4096
#define CHW  (CC * HW)   // 16384

typedef __bf16 v16bf __attribute__((ext_vector_type(16)));
typedef float  v8f   __attribute__((ext_vector_type(8)));

union FragU { v16bf v; uint4 q[2]; };

__device__ __forceinline__ unsigned short f32_to_bf16_rne(float f) {
  union { float f; unsigned int u; } v; v.f = f;
  unsigned int u = v.u;
  unsigned int r = u + 0x7FFFu + ((u >> 16) & 1u);
  return (unsigned short)(r >> 16);
}

// ---------------------------------------------------------------------------
// 1) Signed-edge pooling: one block per edge, f32 atomics into pos/neg pools.
// ---------------------------------------------------------------------------
__global__ __launch_bounds__(256)
void pool_edges_kernel(const float* __restrict__ feats,
                       const long long* __restrict__ edges,
                       float* __restrict__ pos, float* __restrict__ neg) {
  const int e = blockIdx.x;
  const long long src  = edges[3 * e + 0];
  const long long sign = edges[3 * e + 1];
  const long long dst  = edges[3 * e + 2];
  float* pool = (sign > 0) ? pos : neg;
  const float* fs = feats + src * (long long)CHW;
  const float* fd = feats + dst * (long long)CHW;
  float* ps = pool + src * (long long)CHW;
  float* pd = pool + dst * (long long)CHW;
  for (int i = threadIdx.x; i < CHW; i += 256) {
    atomicAdd(&pd[i], fs[i]);
    atomicAdd(&ps[i], fd[i]);
  }
}

// ---------------------------------------------------------------------------
// 2) concat([feats,pos,neg]) + cast -> X1 as (N, HW, 48) bf16.
// ---------------------------------------------------------------------------
__global__ __launch_bounds__(256)
void concat_cast_kernel(const float* __restrict__ feats,
                        const float* __restrict__ posb,
                        const float* __restrict__ negb,
                        unsigned short* __restrict__ X1) {
  long long i = (long long)blockIdx.x * blockDim.x + threadIdx.x;
  const long long total = (long long)NN * HW * 48;
  if (i >= total) return;
  int c = (int)(i % 48);
  long long t = i / 48;
  int p = (int)(t % HW);
  int n = (int)(t / HW);
  float val;
  if (c < 16)      val = feats[((long long)n * 16 + c)        * HW + p];
  else if (c < 32) val = posb [((long long)n * 16 + (c - 16)) * HW + p];
  else             val = negb [((long long)n * 16 + (c - 32)) * HW + p];
  X1[i] = f32_to_bf16_rne(val);
}

// ---------------------------------------------------------------------------
// 3) Weight cast f32 -> bf16, [tap][Cout][CIPAD], channel-padded with zeros.
// ---------------------------------------------------------------------------
__global__ __launch_bounds__(256)
void weight_cast_kernel(const float* __restrict__ w, unsigned short* __restrict__ wb,
                        int cout, int cin, int cipad) {
  int i = blockIdx.x * blockDim.x + threadIdx.x;
  if (i >= 9 * cout * cipad) return;
  int ci  = i % cipad;
  int t2  = i / cipad;
  int co  = t2 % cout;
  int tap = t2 / cout;
  wb[i] = (ci < cin) ? f32_to_bf16_rne(w[((long long)co * cin + ci) * 9 + tap])
                     : (unsigned short)0;
}

// ---------------------------------------------------------------------------
// 4) Tap-decomposed implicit-GEMM conv, one workgroup (8 wave32) per image.
//    LDS: XT[pos][XROW] bf16 + zero row + W[tap][co][CIPAD].
//    Fragments: 2x ds_load_b128 each, double-buffered across chunks.
// ---------------------------------------------------------------------------
template <int CIN, int CIPAD, int COUT, bool OUT_F32>
__global__ __launch_bounds__(256)
void conv3x3_wmma_kernel(const unsigned short* __restrict__ Xin, // (N,HW,CIN) bf16
                         const unsigned short* __restrict__ Wb,  // [9][COUT][CIPAD] bf16
                         const float* __restrict__ bias,         // (COUT)
                         void* __restrict__ Out) {
  constexpr int XROW   = CIPAD + 8;      // row stride (ushorts), 16B aligned
  constexpr int NWAVES = 8;
  constexpr int KCH    = CIPAD / 32;     // K chunks per tap
  constexpr int NCHK   = 9 * KCH;        // total chunks per tile
  extern __shared__ unsigned short lds[];
  unsigned short* ldsXT   = lds;                       // HW * XROW
  unsigned short* ldsZero = lds + HW * XROW;           // XROW (zeroed)
  unsigned short* ldsW    = ldsZero + XROW;            // 9*COUT*CIPAD

  const int n   = blockIdx.x;
  const int tid = threadIdx.x;
  const unsigned short* xg = Xin + (long long)n * HW * CIN;

  // ---- stage input plane (coalesced 16B copies)
  constexpr int RCH = CIN / 8;
  for (int idx = tid; idx < HW * RCH; idx += 256) {
    int p = idx / RCH, s = idx - p * RCH;
    *(uint4*)(ldsXT + p * XROW + s * 8) = *(const uint4*)(xg + p * CIN + s * 8);
  }
  if (CIPAD > CIN) {
    constexpr int PCH = (CIPAD - CIN) / 8;
    for (int idx = tid; idx < HW * PCH; idx += 256) {
      int p = idx / PCH, s = idx - p * PCH;
      *(uint4*)(ldsXT + p * XROW + CIN + s * 8) = make_uint4(0, 0, 0, 0);
    }
  }
  for (int i = tid; i < XROW; i += 256) ldsZero[i] = 0;
  for (int i = tid; i < 9 * COUT * CIPAD / 8; i += 256)
    *(uint4*)(ldsW + i * 8) = *(const uint4*)(Wb + i * 8);
  __syncthreads();

  const int lane  = tid & 31;
  const int wave  = tid >> 5;
  const int row16 = lane & 15;
  const int half  = lane >> 4;
  constexpr int NT  = HW / 16;     // 64 spatial tiles
  constexpr int MT  = COUT / 16;
  constexpr int TOT = MT * NT;

#pragma unroll 1
  for (int t = wave; t < TOT; t += NWAVES) {   // wave-uniform -> EXEC all-1 at WMMA
    const int mt = t / NT;
    const int nt = t - mt * NT;
    const int pos0 = nt * 16 + row16;
    const int oy = pos0 >> 5;
    const int ox = pos0 & 31;

    // per-tap row pointers (one v_cndmask each), hoisted out of the chunk loop
    const unsigned short* rowp[9];
#pragma unroll
    for (int tap = 0; tap < 9; ++tap) {
      const int dy = tap / 3, dx = tap % 3;
      const int iy = oy + dy - 1;
      const int ix = ox + dx - 1;
      const bool inb = ((unsigned)iy < 32u) & ((unsigned)ix < 32u);
      rowp[tap] = inb ? (ldsXT + (iy * 32 + ix) * XROW) : ldsZero;
    }
    const unsigned short* wbase = ldsW + (mt * 16 + row16) * CIPAD;

    // fragment loaders: chunk j = tap*KCH + kc (indices fold at compile time)
    auto loadA = [&](int j, FragU& a) {
      const int tap = j / KCH, kc = j - tap * KCH;
      const unsigned short* wp = wbase + tap * (COUT * CIPAD) + kc * 32;
      a.q[0] = *(const uint4*)(wp + half * 8);
      a.q[1] = *(const uint4*)(wp + 16 + half * 8);
    };
    auto loadB = [&](int j, FragU& b) {
      const int tap = j / KCH, kc = j - tap * KCH;
      const unsigned short* xp = rowp[tap] + kc * 32 + half * 8;
      b.q[0] = *(const uint4*)(xp);
      b.q[1] = *(const uint4*)(xp + 16);
    };

    // double-buffered chunk pipeline: issue loads for j+1 before WMMA of j
    v8f acc = {};
    FragU af[2], bf[2];
    loadA(0, af[0]);
    loadB(0, bf[0]);
#pragma unroll
    for (int j = 0; j < NCHK; ++j) {
      if (j + 1 < NCHK) {
        loadA(j + 1, af[(j + 1) & 1]);
        loadB(j + 1, bf[(j + 1) & 1]);
      }
      acc = __builtin_amdgcn_wmma_f32_16x16x32_bf16(
          false, af[j & 1].v, false, bf[j & 1].v, (short)0, acc, false, false);
    }

    // ---- epilogue: bias + leaky ReLU(0.1)
    const int co0 = mt * 16 + half * 8;
    if (OUT_F32) {
#pragma unroll
      for (int r = 0; r < 8; ++r) {
        float y = acc[r] + bias[co0 + r];
        y = (y > 0.0f) ? y : 0.1f * y;
        ((float*)Out)[((long long)n * COUT + co0 + r) * HW + pos0] = y;
      }
    } else {
      union { uint4 q; unsigned short s[8]; } ov;
#pragma unroll
      for (int r = 0; r < 8; ++r) {
        float y = acc[r] + bias[co0 + r];
        y = (y > 0.0f) ? y : 0.1f * y;
        ov.s[r] = f32_to_bf16_rne(y);
      }
      *(uint4*)((unsigned short*)Out + ((long long)n * HW + pos0) * COUT + co0) = ov.q;
    }
  }
}

// ---------------------------------------------------------------------------
// Host driver
// ---------------------------------------------------------------------------
extern "C" void kernel_launch(void* const* d_in, const int* in_sizes, int n_in,
                              void* d_out, int out_size, void* d_ws, size_t ws_size,
                              hipStream_t stream) {
  (void)in_sizes; (void)n_in; (void)out_size; (void)ws_size;

  const float*     feats = (const float*)d_in[0];
  const long long* edges = (const long long*)d_in[1];
  const float*     w1    = (const float*)d_in[2];
  const float*     b1    = (const float*)d_in[3];
  const float*     w2    = (const float*)d_in[4];
  const float*     b2    = (const float*)d_in[5];
  const float*     w3    = (const float*)d_in[6];
  const float*     b3    = (const float*)d_in[7];

  char* ws = (char*)d_ws;
  const size_t SZF = (size_t)NN * CHW * sizeof(float);      // 64 MB per pool
  float*          posb = (float*)(ws);
  float*          negb = (float*)(ws + SZF);
  unsigned short* X1   = (unsigned short*)(ws + 2 * SZF);   // (N,HW,48) bf16
  unsigned short* Y1   = (unsigned short*)(ws);             // overlay pos
  unsigned short* Y2   = (unsigned short*)(ws + SZF);       // overlay neg
  size_t woff = 2 * SZF + (size_t)NN * HW * 48 * 2;
  unsigned short* Wb1 = (unsigned short*)(ws + woff);       // 9*32*64
  unsigned short* Wb2 = Wb1 + 9 * 32 * 64;                  // 9*32*32
  unsigned short* Wb3 = Wb2 + 9 * 32 * 32;                  // 9*16*32

  hipMemsetAsync(posb, 0, 2 * SZF, stream);
  pool_edges_kernel<<<EE, 256, 0, stream>>>(feats, edges, posb, negb);

  {
    long long tot = (long long)NN * HW * 48;
    concat_cast_kernel<<<(int)((tot + 255) / 256), 256, 0, stream>>>(feats, posb, negb, X1);
  }

  weight_cast_kernel<<<(9 * 32 * 64 + 255) / 256, 256, 0, stream>>>(w1, Wb1, 32, 48, 64);
  weight_cast_kernel<<<(9 * 32 * 32 + 255) / 256, 256, 0, stream>>>(w2, Wb2, 32, 32, 32);
  weight_cast_kernel<<<(9 * 16 * 32 + 255) / 256, 256, 0, stream>>>(w3, Wb3, 16, 32, 32);

  {
    size_t l1 = (size_t)(HW * 72 + 72 + 9 * 32 * 64) * 2;   // ~184 KB
    size_t l2 = (size_t)(HW * 40 + 40 + 9 * 32 * 32) * 2;   // ~100 KB
    size_t l3 = (size_t)(HW * 40 + 40 + 9 * 16 * 32) * 2;   // ~91 KB
    conv3x3_wmma_kernel<48, 64, 32, false><<<NN, 256, l1, stream>>>(X1, Wb1, b1, Y1);
    conv3x3_wmma_kernel<32, 32, 32, false><<<NN, 256, l2, stream>>>(Y1, Wb2, b2, Y2);
    conv3x3_wmma_kernel<32, 32, 16, true ><<<NN, 256, l3, stream>>>(Y2, Wb3, b3, d_out);
  }
}
